// SelfAttention_18245021073428
// MI455X (gfx1250) — matrix operand order repaired and checked by reading, not compiled, not run on hardware
//
#include <hip/hip_runtime.h>
#include <hip/hip_bf16.h>

typedef __bf16 bf16;
typedef __attribute__((ext_vector_type(16))) __bf16 v16bf;
typedef __attribute__((ext_vector_type(8)))  __bf16 v8bf;
typedef __attribute__((ext_vector_type(8)))  float  v8f;

#define S_LEN   4096
#define DMODEL  1024
#define DHEAD   128

// ---------- WMMA fragment loaders (contiguous k = 16*half + e in both A and B) ----------

// A fragment, 16x32 bf16, source row-major [rows, ld].
// lane<16 holds row M=lane, K = 0..15; lane>=16 same row, K = 16..31.
__device__ __forceinline__ v16bf load_a_rowmajor(const bf16* base, int ld, int row_base, int k_base) {
    const int lane = threadIdx.x & 31;
    const int half = lane >> 4;
    const int m    = lane & 15;
    const bf16* p = base + (size_t)(row_base + m) * ld + k_base + half * 16;
    v8bf lo = *(const v8bf*)(p);        // K = half*16 + 0..7
    v8bf hi = *(const v8bf*)(p + 8);    // K = half*16 + 8..15
    v16bf a;
#pragma unroll
    for (int i = 0; i < 8; ++i) { a[i] = lo[i]; a[i + 8] = hi[i]; }
    return a;
}

// B fragment, 32x16 bf16, source "K-contiguous": value(k,n) = baseT[(n_base+n)*ldT + k_base + k].
// lane holds column n = lane&15; element e -> K = 16*half + e (contiguous 32 bytes).
__device__ __forceinline__ v16bf load_b_kcontig(const bf16* baseT, int ldT, int n_base, int k_base) {
    const int lane = threadIdx.x & 31;
    const int half = lane >> 4;
    const int n    = lane & 15;
    const bf16* p = baseT + (size_t)(n_base + n) * ldT + k_base + half * 16;
    v8bf lo = *(const v8bf*)(p);
    v8bf hi = *(const v8bf*)(p + 8);
    v16bf b;
#pragma unroll
    for (int i = 0; i < 8; ++i) { b[i] = lo[i]; b[i + 8] = hi[i]; }
    return b;
}

__device__ __forceinline__ v8f wmma_bf16(v16bf a, v16bf b, v8f c) {
    return __builtin_amdgcn_wmma_f32_16x16x32_bf16(false, a, false, b, (short)0, c, false, false);
}

// ---------- fp32 -> bf16 conversion ----------
__global__ void cvt_f32_bf16_kernel(const float* __restrict__ in, bf16* __restrict__ out, int n) {
    int i = blockIdx.x * blockDim.x + threadIdx.x;
    if (i < n) out[i] = (bf16)in[i];
}

// ---------- QKV projection: one wave per 16x16 tile of [S, 128] ----------
// which = blockIdx.z: 0 -> Q (row-major), 1 -> K (row-major), 2 -> V (stored transposed Vt[128][S])
__global__ void proj_kernel(const bf16* __restrict__ Xb,
                            const bf16* __restrict__ Wq, const bf16* __restrict__ Wk,
                            const bf16* __restrict__ Wv,
                            const float* __restrict__ bq, const float* __restrict__ bk,
                            const float* __restrict__ bv,
                            bf16* __restrict__ Qb, bf16* __restrict__ Kb, bf16* __restrict__ Vt) {
    const int mt = blockIdx.x;           // 0..255  (seq tiles)
    const int nt = blockIdx.y;           // 0..7    (feature tiles)
    const int which = blockIdx.z;        // 0..2
    const bf16*  W    = (which == 0) ? Wq : (which == 1) ? Wk : Wv;
    const float* bias = (which == 0) ? bq : (which == 1) ? bk : bv;

    v8f c = {};
    for (int k = 0; k < DMODEL; k += 32) {
        v16bf a = load_a_rowmajor(Xb, DMODEL, mt * 16, k);       // X[s, k]
        v16bf b = load_b_kcontig(W, DMODEL, nt * 16, k);         // W[n, k] (W^T as B)
        c = wmma_bf16(a, b, c);
    }

    const int lane = threadIdx.x & 31;
    const int half = lane >> 4;
    const int ncol = lane & 15;
    const float bval = bias[nt * 16 + ncol];
#pragma unroll
    for (int r = 0; r < 8; ++r) {
        const int row = mt * 16 + half * 8 + r;                  // seq index
        const float v = c[r] + bval;
        if (which == 2) {
            Vt[(size_t)(nt * 16 + ncol) * S_LEN + row] = (bf16)v;
        } else {
            bf16* dst = (which == 0) ? Qb : Kb;
            dst[(size_t)row * DHEAD + nt * 16 + ncol] = (bf16)v;
        }
    }
}

// ---------- Flash attention: one wave per 16 query rows ----------
__global__ void attn_kernel(const bf16* __restrict__ Qb, const bf16* __restrict__ Kb,
                            const bf16* __restrict__ Vt, float* __restrict__ out) {
    __shared__ alignas(16) bf16 Pshm[16 * 32];   // P tile staging (C-layout -> A-layout)

    const int qt   = blockIdx.x;                  // query tile 0..255
    const int lane = threadIdx.x & 31;
    const int half = lane >> 4;
    const int ncol = lane & 15;

    // Q fragments: 16 rows x 128 dims = 4 A-fragments (resident)
    v16bf aq[4];
#pragma unroll
    for (int kk = 0; kk < 4; ++kk)
        aq[kk] = load_a_rowmajor(Qb, DHEAD, qt * 16, kk * 32);

    v8f o[8] = {};                                // O accumulator: 16 x 128 f32
    float mrow[8], lrow[8];
#pragma unroll
    for (int r = 0; r < 8; ++r) { mrow[r] = -3.0e38f; lrow[r] = 0.0f; }

    const float sc = 1.0f / 128.0f;               // reference scales by d_qk

    for (int j = 0; j < S_LEN; j += 32) {
        // scores: two 16x16 tiles (keys j..j+15, j+16..j+31), K-dot over 128
        v8f c0 = {}, c1 = {};
#pragma unroll
        for (int kk = 0; kk < 4; ++kk) {
            v16bf b0 = load_b_kcontig(Kb, DHEAD, j,      kk * 32);
            v16bf b1 = load_b_kcontig(Kb, DHEAD, j + 16, kk * 32);
            c0 = wmma_bf16(aq[kk], b0, c0);
            c1 = wmma_bf16(aq[kk], b1, c1);
        }

        __syncthreads();   // protect Pshm: previous iteration's reads done
#pragma unroll
        for (int r = 0; r < 8; ++r) {
            float s0 = c0[r] * sc;
            float s1 = c1[r] * sc;
            // row max across the 16 lanes of this half (xor masks < 16 stay in half)
            float t = fmaxf(s0, s1);
#pragma unroll
            for (int off = 8; off >= 1; off >>= 1) t = fmaxf(t, __shfl_xor(t, off, 32));
            const float mnew  = fmaxf(mrow[r], t);
            const float alpha = __expf(mrow[r] - mnew);
            const float p0 = __expf(s0 - mnew);
            const float p1 = __expf(s1 - mnew);
            float rs = p0 + p1;
#pragma unroll
            for (int off = 8; off >= 1; off >>= 1) rs += __shfl_xor(rs, off, 32);
            lrow[r] = lrow[r] * alpha + rs;
            mrow[r] = mnew;
#pragma unroll
            for (int f = 0; f < 8; ++f) o[f][r] *= alpha;
            // stash P in C-layout, bf16
            const int row = half * 8 + r;
            Pshm[row * 32 + ncol]      = (bf16)p0;
            Pshm[row * 32 + 16 + ncol] = (bf16)p1;
        }
        __syncthreads();

        // reload P as A fragment (16x32, row-major in LDS, ld=32) -> ds_load_b128 x2
        v16bf ap;
        {
            const int m = lane & 15;
            v8bf lo = *(const v8bf*)(&Pshm[m * 32 + half * 16]);
            v8bf hi = *(const v8bf*)(&Pshm[m * 32 + half * 16 + 8]);
#pragma unroll
            for (int i = 0; i < 8; ++i) { ap[i] = lo[i]; ap[i + 8] = hi[i]; }
        }

        // O += P @ V : B = Vt (value-dim major), k = key index (contiguous)
#pragma unroll
        for (int f = 0; f < 8; ++f) {
            v16bf bv = load_b_kcontig(Vt, S_LEN, f * 16, j);
            o[f] = wmma_bf16(ap, bv, o[f]);
        }
    }

    // normalize and write out fp32 [S, 128]
#pragma unroll
    for (int r = 0; r < 8; ++r) {
        const float inv = 1.0f / lrow[r];
        const int row = qt * 16 + half * 8 + r;
#pragma unroll
        for (int f = 0; f < 8; ++f)
            out[(size_t)row * DHEAD + f * 16 + ncol] = o[f][r] * inv;
    }
}

// ---------- launch ----------
extern "C" void kernel_launch(void* const* d_in, const int* in_sizes, int n_in,
                              void* d_out, int out_size, void* d_ws, size_t ws_size,
                              hipStream_t stream) {
    const float* X  = (const float*)d_in[0];
    const float* qW = (const float*)d_in[1];
    const float* qb = (const float*)d_in[2];
    const float* kW = (const float*)d_in[3];
    const float* kb = (const float*)d_in[4];
    const float* vW = (const float*)d_in[5];
    const float* vb = (const float*)d_in[6];
    float* out = (float*)d_out;

    char* ws = (char*)d_ws;
    size_t off = 0;
    bf16* Xb  = (bf16*)(ws + off); off += (size_t)S_LEN * DMODEL * 2;   // 8 MiB
    bf16* Wqb = (bf16*)(ws + off); off += (size_t)DHEAD * DMODEL * 2;
    bf16* Wkb = (bf16*)(ws + off); off += (size_t)DHEAD * DMODEL * 2;
    bf16* Wvb = (bf16*)(ws + off); off += (size_t)DHEAD * DMODEL * 2;
    bf16* Qb  = (bf16*)(ws + off); off += (size_t)S_LEN * DHEAD * 2;
    bf16* Kb  = (bf16*)(ws + off); off += (size_t)S_LEN * DHEAD * 2;
    bf16* Vt  = (bf16*)(ws + off); off += (size_t)S_LEN * DHEAD * 2;    // total ~11.8 MiB

    // 1) fp32 -> bf16
    {
        int nX = S_LEN * DMODEL;
        cvt_f32_bf16_kernel<<<(nX + 255) / 256, 256, 0, stream>>>(X, Xb, nX);
        int nW = DHEAD * DMODEL;
        cvt_f32_bf16_kernel<<<(nW + 255) / 256, 256, 0, stream>>>(qW, Wqb, nW);
        cvt_f32_bf16_kernel<<<(nW + 255) / 256, 256, 0, stream>>>(kW, Wkb, nW);
        cvt_f32_bf16_kernel<<<(nW + 255) / 256, 256, 0, stream>>>(vW, Wvb, nW);
    }

    // 2) Q, K (row-major) and V (transposed) projections via WMMA
    {
        dim3 grid(S_LEN / 16, DHEAD / 16, 3);
        proj_kernel<<<grid, 32, 0, stream>>>(Xb, Wqb, Wkb, Wvb, qb, kb, vb, Qb, Kb, Vt);
    }

    // 3) flash attention, one wave per 16 query rows
    attn_kernel<<<S_LEN / 16, 32, 0, stream>>>(Qb, Kb, Vt, out);
}
